// PSAGNNNet_73452530696951
// MI455X (gfx1250) — matrix operand order repaired and checked by reading, not compiled, not run on hardware
//
#include <hip/hip_runtime.h>
#include <hip/hip_bf16.h>

// ============================================================================
// PSAGNN for MI455X (gfx1250, wave32, WMMA).
//
// Design notes (no runtime available; reasoned from MI455X specs):
//  * Dense GEMM work (~6 GFLOP total) is negligible vs. WMMA throughput; the
//    kernel set is bound by irregular edge traffic (~2-4 GB over 7 edge
//    passes -> O(100us) at 23.3 TB/s HBM; node-state buffers (<45 MB) fit in
//    the 192 MB L2 so edge atomics resolve in-cache).
//  * All dense projections use v_wmma_f32_16x16x32_f16 (codegen-confirmed
//    builtin). Features are kept in f16 to halve gather bytes; accumulation
//    is f32. Weights are pre-transposed to column-major f16 so each lane's
//    packed K-pair of the B fragment is one contiguous 32-bit load.
//  * Fragment layouts follow CDNA5 ISA 7.12.2 (16-bit A 16x32:
//    lane&15 = M row, lane>>4 selects K 8..15 / 24..31 halves; 32-bit C/D:
//    lane&15 = N, VGPR r -> M = r + 8*(lane>>4)).
//  * Gumbel top-k: JAX's RNG stream is not reproducible here; a deterministic
//    integer-hash Gumbel stands in. Selection is a 1024-bin histogram
//    threshold-select (strictly-above bins all kept, threshold bin fills the
//    remaining quota) -- one O(E) pass instead of a sort.
//  * Segment ops (seg_sum/seg_max) use atomics; float max uses an
//    order-preserving uint encoding so atomicMax works for negatives.
// ============================================================================

static constexpr int       kN    = 10000;
static constexpr int       kF    = 32;
static constexpr int       kH    = 64;
static constexpr int       kL    = 10;
static constexpr int       kLP   = 16;                 // L padded to one WMMA tile
static constexpr int       kE    = 320000;
static constexpr int       kR    = 4;
static constexpr int       kNB   = kR * kN;            // 40000 batched nodes
static constexpr long long kRE   = (long long)kR * kE; // 1,280,000 batched edges
static constexpr int       kNRES = (70 * kE) / 100;    // 224000 kept edges / run
static constexpr int       kBIN  = 1024;

typedef __attribute__((ext_vector_type(16))) _Float16 v16h;
typedef __attribute__((ext_vector_type(8)))  float    v8f;

// ---------------- device helpers ----------------
__device__ __forceinline__ unsigned f2u(float f) {
  unsigned u = __float_as_uint(f);
  return (u & 0x80000000u) ? ~u : (u | 0x80000000u);   // order-preserving map
}
__device__ __forceinline__ float u2f(unsigned u) {
  unsigned o = (u & 0x80000000u) ? (u ^ 0x80000000u) : ~u;
  return __uint_as_float(o);
}
__device__ __forceinline__ float leaky02(float x) { return x > 0.f ? x : 0.2f * x; }
__device__ __forceinline__ float gumbel_hash(unsigned a) {
  a ^= a >> 16; a *= 0x7feb352du; a ^= a >> 15; a *= 0x846ca68bu; a ^= a >> 16;
  float u = ((a >> 8) + 0.5f) * (1.0f / 16777216.0f);  // (0,1)
  return -logf(-logf(u));
}
__device__ __forceinline__ void edge_nodes(const long long* ei, long long i,
                                           int& sN, int& dN) {
  int r = (int)(i / kE);
  int e = (int)(i - (long long)r * kE);
  sN = r * kN + (int)ei[e];
  dN = r * kN + (int)ei[kE + e];
}

#define GRID_STRIDE(i, total)                                                   \
  for (long long i = (long long)blockIdx.x * blockDim.x + threadIdx.x;          \
       i < (long long)(total); i += (long long)gridDim.x * blockDim.x)

// ---------------- stage 0: column standardize (ddof=1) ----------------
__global__ void psagnn_colstats(const float* __restrict__ x,
                                float* __restrict__ mean, float* __restrict__ rstd) {
  __shared__ float s1[256], s2[256];
  int f = blockIdx.x, t = threadIdx.x;
  float a = 0.f, b = 0.f;
  for (int n = t; n < kN; n += 256) {
    float v = x[(long long)n * kF + f];
    a += v; b += v * v;
  }
  s1[t] = a; s2[t] = b; __syncthreads();
  for (int o = 128; o > 0; o >>= 1) {
    if (t < o) { s1[t] += s1[t + o]; s2[t] += s2[t + o]; }
    __syncthreads();
  }
  if (t == 0) {
    float m = s1[0] / (float)kN;
    float var = (s2[0] - (float)kN * m * m) / (float)(kN - 1);
    mean[f] = m;
    rstd[f] = rsqrtf(fmaxf(var, 1e-20f));
  }
}

// xr[r,n,f] = drop ? 0 : (x-mean)*rstd, stored f16 (halves gather bytes, WMMA-ready)
__global__ void psagnn_build_xr(const float* __restrict__ x,
                                const unsigned char* __restrict__ drop,
                                const float* __restrict__ mean,
                                const float* __restrict__ rstd,
                                _Float16* __restrict__ xrh) {
  GRID_STRIDE(i, (long long)kNB * kF) {
    int node = (int)(i / kF), f = (int)(i % kF);
    int n = node % kN;
    float v = drop[node] ? 0.f : (x[(long long)n * kF + f] - mean[f]) * rstd[f];
    xrh[i] = (_Float16)v;
  }
}

// column-major (B^T) f16 weight copies; W2 padded 10 -> 16 columns with zeros
__global__ void psagnn_prep_weights(const float* __restrict__ W1,  // [32,64]
                                    const float* __restrict__ Wg,  // [64,64]
                                    const float* __restrict__ W2,  // [64,10]
                                    const float* __restrict__ Wm1, // [32,64]
                                    _Float16* __restrict__ W1c,    // [64][32]
                                    _Float16* __restrict__ Wgc,    // [64][64]
                                    _Float16* __restrict__ W2c,    // [16][64]
                                    _Float16* __restrict__ Wm1c) { // [64][32]
  int t = threadIdx.x;
  for (int i = t; i < 64 * 32; i += 256) { int c = i / 32, k = i % 32; W1c[i]  = (_Float16)W1[k * 64 + c]; }
  for (int i = t; i < 64 * 64; i += 256) { int c = i / 64, k = i % 64; Wgc[i]  = (_Float16)Wg[k * 64 + c]; }
  for (int i = t; i < 16 * 64; i += 256) { int c = i / 64, k = i % 64; W2c[i]  = (_Float16)((c < kL) ? W2[k * kL + c] : 0.f); }
  for (int i = t; i < 64 * 32; i += 256) { int c = i / 32, k = i % 32; Wm1c[i] = (_Float16)Wm1[k * 64 + c]; }
}

// ---------------- stage 1: edge-score MLP, fully fused, WMMA ----------------
// One wave = 16 edges. A = (xr[src]-xr[dst]) 16x32 f16, 4 WMMAs vs Wm1 tiles,
// fused bias+relu+dot(Wm2) in registers, shfl_xor reduction, write s[r,e].
__global__ void psagnn_edge_mlp_wmma(const _Float16* __restrict__ xrh,
                                     const long long* __restrict__ ei,
                                     const _Float16* __restrict__ Wm1c,
                                     const float* __restrict__ bm1,
                                     const float* __restrict__ Wm2,
                                     const float* __restrict__ bm2,
                                     float* __restrict__ s) {
  int wave = (int)((blockIdx.x * blockDim.x + threadIdx.x) >> 5);
  int lane = threadIdx.x & 31;
  const int TPR = kE / 16;                 // 20000 tiles per run (exact)
  int r = wave / TPR;
  if (r >= kR) return;                     // wave-uniform
  int tile = wave - r * TPR;
  int l15 = lane & 15, hi = lane >> 4;

  int e = tile * 16 + l15;
  const _Float16* xa = xrh + (long long)(r * kN + (int)ei[e])       * kF;
  const _Float16* xb = xrh + (long long)(r * kN + (int)ei[kE + e])  * kF;

  v16h A;
#pragma unroll
  for (int v = 0; v < 8; ++v) {            // ISA 16-bit A 16x32 layout
    int ka = ((v < 4) ? 0 : 16) + hi * 8 + (v & 3) * 2;
    A[2 * v]     = (_Float16)(xa[ka]     - xb[ka]);
    A[2 * v + 1] = (_Float16)(xa[ka + 1] - xb[ka + 1]);
  }

  float sacc[8];
#pragma unroll
  for (int i = 0; i < 8; ++i) sacc[i] = 0.f;

#pragma unroll
  for (int t = 0; t < 4; ++t) {            // 64 hidden cols = 4 N-tiles
    int col = t * 16 + l15;
    const _Float16* bp = Wm1c + (long long)col * kF;
    v16h B;
#pragma unroll
    for (int v = 0; v < 8; ++v) {          // B 32x16: same K order as A (lanes 0-15 K=0..7,16..23; 16-31 K=8..15,24..31)
      int kb = ((v < 4) ? 0 : 16) + hi * 8 + (v & 3) * 2;
      B[2 * v] = bp[kb]; B[2 * v + 1] = bp[kb + 1];
    }
    v8f c = {};
    c = __builtin_amdgcn_wmma_f32_16x16x32_f16(false, A, false, B,
                                               (short)0, c, false, false);
    float b1v = bm1[col], w2v = Wm2[col];
#pragma unroll
    for (int rr = 0; rr < 8; ++rr)
      sacc[rr] += fmaxf(c[rr] + b1v, 0.f) * w2v;   // relu + second layer
  }
  // reduce over the 16 N-lanes (xor<16 keeps hi-half intact)
#pragma unroll
  for (int off = 1; off < 16; off <<= 1)
#pragma unroll
    for (int rr = 0; rr < 8; ++rr)
      sacc[rr] += __shfl_xor(sacc[rr], off, 32);
  if (l15 == 0) {
    float b2 = bm2[0];
#pragma unroll
    for (int rr = 0; rr < 8; ++rr)
      s[(long long)r * kE + tile * 16 + hi * 8 + rr] = sacc[rr] + b2;
  }
}

// ---------------- stage 2: min/max, keys, histogram top-k select ------------
__global__ void psagnn_init_control(unsigned* mmin, unsigned* mmax, unsigned* hist) {
  int i = blockIdx.x * blockDim.x + threadIdx.x;
  if (i < kR) { mmin[i] = 0xFFFFFFFFu; mmax[i] = 0u; }
  for (int j = i; j < kR * kBIN; j += gridDim.x * blockDim.x) hist[j] = 0u;
}

__global__ void psagnn_minmax(const float* __restrict__ s,
                              unsigned* __restrict__ mmin, unsigned* __restrict__ mmax) {
  GRID_STRIDE(i, kRE) {
    int r = (int)(i / kE);
    unsigned u = f2u(s[i]);
    atomicMin(&mmin[r], u);
    atomicMax(&mmax[r], u);
  }
}

__global__ void psagnn_key_hist(const float* __restrict__ s,
                                const unsigned* __restrict__ mmin,
                                const unsigned* __restrict__ mmax,
                                unsigned* __restrict__ keyu, unsigned* __restrict__ hist) {
  GRID_STRIDE(i, kRE) {
    int r = (int)(i / kE);
    float mn = u2f(mmin[r]), mx = u2f(mmax[r]);
    float d = mx - mn;
    float w = (d > 0.f) ? (s[i] - mn) / d : 0.f;
    float key = logf(fmaxf(w, 1e-38f)) + gumbel_hash((unsigned)i * 2654435761u + 0x9e3779b9u);
    unsigned ku = f2u(key);
    keyu[i] = ku;
    atomicAdd(&hist[r * kBIN + (ku >> 22)], 1u);
  }
}

// thr[r] = {threshold bin T, count strictly above, quota within T}
__global__ void psagnn_threshold(const unsigned* __restrict__ hist,
                                 unsigned* __restrict__ thr, unsigned* __restrict__ cnt) {
  int r = blockIdx.x;
  if (threadIdx.x == 0) {
    unsigned cum = 0, above = 0; int T = 0;
    for (int b = kBIN - 1; b >= 0; --b) {
      unsigned c = hist[r * kBIN + b];
      if (cum + c >= (unsigned)kNRES) { T = b; above = cum; break; }
      cum += c;
    }
    thr[r * 3 + 0] = (unsigned)T;
    thr[r * 3 + 1] = above;
    thr[r * 3 + 2] = (unsigned)kNRES - above;
    cnt[r * 2 + 0] = 0u;
    cnt[r * 2 + 1] = 0u;
  }
}

__global__ void psagnn_select(const unsigned* __restrict__ keyu,
                              const unsigned* __restrict__ thr,
                              unsigned* __restrict__ cnt, int* __restrict__ sel) {
  GRID_STRIDE(i, kRE) {
    int r = (int)(i / kE);
    int e = (int)(i - (long long)r * kE);
    unsigned b = keyu[i] >> 22;
    unsigned T = thr[r * 3], above = thr[r * 3 + 1], quota = thr[r * 3 + 2];
    if (b > T) {
      unsigned p = atomicAdd(&cnt[r * 2], 1u);
      if (p < (unsigned)kNRES) sel[(long long)r * kNRES + p] = e;
    } else if (b == T) {
      unsigned q = atomicAdd(&cnt[r * 2 + 1], 1u);
      if (q < quota) sel[(long long)r * kNRES + above + q] = e;
    }
  }
}

// second output: kept edge endpoints as floats (harness output dtype)
__global__ void psagnn_write_sel(const int* __restrict__ sel,
                                 const long long* __restrict__ ei,
                                 float* __restrict__ out) {
  GRID_STRIDE(j, (long long)kR * kNRES) {
    int r = (int)(j / kNRES), k = (int)(j % kNRES);
    int e = sel[j];
    out[(long long)r * 2 * kNRES + k]         = (float)ei[e];
    out[(long long)r * 2 * kNRES + kNRES + k] = (float)ei[kE + e];
  }
}

// ---------------- stage 3: generic f16 WMMA GEMM  C[M,Nn] = A[M,K] @ B ------
// A row-major f16; Bc column-major f16 ([Nn][K]); one wave per 16x16 tile.
__global__ void psagnn_gemm_wmma(const _Float16* __restrict__ A,
                                 const _Float16* __restrict__ Bc,
                                 float* __restrict__ C, int M, int K, int Nn) {
  int wave = (int)((blockIdx.x * blockDim.x + threadIdx.x) >> 5);
  int lane = threadIdx.x & 31;
  int tilesN = Nn >> 4;
  int tiles = (M >> 4) * tilesN;
  if (wave >= tiles) return;               // wave-uniform -> EXEC all-1 for WMMA
  int tm = (wave / tilesN) << 4;
  int tn = (wave % tilesN) << 4;
  int l15 = lane & 15, hi = lane >> 4;
  v8f c = {};
  for (int k0 = 0; k0 < K; k0 += 32) {
    v16h Af, Bf;
    const _Float16* ap = A + (long long)(tm + l15) * K + k0;
#pragma unroll
    for (int v = 0; v < 8; ++v) {
      int ka = ((v < 4) ? 0 : 16) + hi * 8 + (v & 3) * 2;
      Af[2 * v] = ap[ka]; Af[2 * v + 1] = ap[ka + 1];
    }
    const _Float16* bp = Bc + (long long)(tn + l15) * K + k0;
#pragma unroll
    for (int v = 0; v < 8; ++v) {
      int kb = ((v < 4) ? 0 : 16) + hi * 8 + (v & 3) * 2;
      Bf[2 * v] = bp[kb]; Bf[2 * v + 1] = bp[kb + 1];
    }
    c = __builtin_amdgcn_wmma_f32_16x16x32_f16(false, Af, false, Bf,
                                               (short)0, c, false, false);
  }
  float* cp = C + (long long)(tm + hi * 8) * Nn + tn + l15;
#pragma unroll
  for (int rr = 0; rr < 8; ++rr) cp[(long long)rr * Nn] = c[rr];
}

// ---------------- stage 4: GCN aggregate (shared degree, atomics) -----------
__global__ void psagnn_deg_init(float* deg) {
  GRID_STRIDE(i, kNB) deg[i] = 1.0f;       // self loop
}
__global__ void psagnn_deg_edge(const long long* __restrict__ ei, float* __restrict__ deg) {
  GRID_STRIDE(i, kRE) {
    int r = (int)(i / kE);
    int e = (int)(i - (long long)r * kE);
    atomicAdd(&deg[r * kN + (int)ei[kE + e]], 1.0f);
  }
}
__global__ void psagnn_dinv(const float* deg, float* dinv) {
  GRID_STRIDE(i, kNB) dinv[i] = rsqrtf(deg[i]);
}
__global__ void psagnn_gcn_init(const float* __restrict__ xw, const float* __restrict__ bias,
                                const float* __restrict__ dinv, float* __restrict__ out,
                                int Out, int biasN) {
  GRID_STRIDE(t, (long long)kNB * Out) {
    int n = (int)(t / Out), o = (int)(t % Out);
    float b = (o < biasN) ? bias[o] : 0.f;
    float di = dinv[n];
    out[t] = b + xw[t] * di * di;          // self-loop term, norm = dinv^2
  }
}
__global__ void psagnn_gcn_edge(const long long* __restrict__ ei,
                                const float* __restrict__ xw, const float* __restrict__ dinv,
                                float* __restrict__ out, int oshift) {
  int Out = 1 << oshift;
  GRID_STRIDE(t, kRE << oshift) {
    long long eidx = t >> oshift;
    int o = (int)(t & (Out - 1));
    int sN, dN; edge_nodes(ei, eidx, sN, dN);
    float c = dinv[sN] * dinv[dN];
    atomicAdd(&out[(long long)dN * Out + o], xw[(long long)sN * Out + o] * c);
  }
}
__global__ void psagnn_relu_h16(const float* __restrict__ in, _Float16* __restrict__ hh,
                                long long n) {
  GRID_STRIDE(i, n) hh[i] = (_Float16)fmaxf(in[i], 0.f);
}

// ---------------- stage 5: GAT attention (seg_max/seg_sum via atomics) ------
__global__ void psagnn_attn_prep(const float* __restrict__ xw,
                                 const float* __restrict__ a_src, const float* __restrict__ a_dst,
                                 float* __restrict__ es, float* __restrict__ ed,
                                 unsigned* __restrict__ mU) {
  GRID_STRIDE(i, kNB) {
    float s = 0.f, d = 0.f;
    const float* p = xw + (long long)i * kH;
#pragma unroll 8
    for (int o = 0; o < kH; ++o) { float v = p[o]; s += v * a_src[o]; d += v * a_dst[o]; }
    es[i] = s; ed[i] = d;
    mU[i] = f2u(leaky02(s + d));           // seg_max init with self-loop value
  }
}
__global__ void psagnn_gat_max(const long long* __restrict__ ei,
                               const float* __restrict__ es, const float* __restrict__ ed,
                               unsigned* __restrict__ mU) {
  GRID_STRIDE(i, kRE) {
    int sN, dN; edge_nodes(ei, i, sN, dN);
    atomicMax(&mU[dN], f2u(leaky02(es[sN] + ed[dN])));
  }
}
__global__ void psagnn_gat_z_init(const float* __restrict__ es, const float* __restrict__ ed,
                                  const unsigned* __restrict__ mU, float* __restrict__ z) {
  GRID_STRIDE(i, kNB) z[i] = expf(leaky02(es[i] + ed[i]) - u2f(mU[i]));
}
__global__ void psagnn_gat_z_edge(const long long* __restrict__ ei,
                                  const float* __restrict__ es, const float* __restrict__ ed,
                                  const unsigned* __restrict__ mU, float* __restrict__ z) {
  GRID_STRIDE(i, kRE) {
    int sN, dN; edge_nodes(ei, i, sN, dN);
    atomicAdd(&z[dN], expf(leaky02(es[sN] + ed[dN]) - u2f(mU[dN])));
  }
}
__global__ void psagnn_gat_out_init(const float* __restrict__ xw, const float* __restrict__ bg,
                                    const float* __restrict__ es, const float* __restrict__ ed,
                                    const unsigned* __restrict__ mU, const float* __restrict__ z,
                                    float* __restrict__ out) {
  GRID_STRIDE(t, (long long)kNB * kH) {
    int n = (int)(t / kH), o = (int)(t % kH);
    float a = expf(leaky02(es[n] + ed[n]) - u2f(mU[n])) / z[n];
    out[t] = bg[o] + xw[t] * a;
  }
}
__global__ void psagnn_gat_out_edge(const long long* __restrict__ ei,
                                    const float* __restrict__ xw,
                                    const float* __restrict__ es, const float* __restrict__ ed,
                                    const unsigned* __restrict__ mU, const float* __restrict__ z,
                                    float* __restrict__ out) {
  GRID_STRIDE(t, kRE * kH) {
    long long eidx = t >> 6;               // kH == 64
    int o = (int)(t & 63);
    int sN, dN; edge_nodes(ei, eidx, sN, dN);
    float alpha = expf(leaky02(es[sN] + ed[dN]) - u2f(mU[dN])) / z[dN];
    atomicAdd(&out[(long long)dN * kH + o], xw[(long long)sN * kH + o] * alpha);
  }
}

// ---------------- stage 6: log_softmax over first 10 padded cols ------------
__global__ void psagnn_logsoftmax(const float* __restrict__ agg, float* __restrict__ out) {
  GRID_STRIDE(i, kNB) {
    float v[kL], mx = -1e30f;
#pragma unroll
    for (int l = 0; l < kL; ++l) { v[l] = agg[(long long)i * kLP + l]; mx = fmaxf(mx, v[l]); }
    float s = 0.f;
#pragma unroll
    for (int l = 0; l < kL; ++l) s += expf(v[l] - mx);
    float ls = logf(s);
#pragma unroll
    for (int l = 0; l < kL; ++l) out[(long long)i * kL + l] = v[l] - mx - ls;
  }
}

// ============================================================================
extern "C" void kernel_launch(void* const* d_in, const int* in_sizes, int n_in,
                              void* d_out, int out_size, void* d_ws, size_t ws_size,
                              hipStream_t stream) {
  (void)in_sizes; (void)n_in; (void)out_size; (void)ws_size;

  const float*         x    = (const float*)d_in[0];
  const long long*     ei   = (const long long*)d_in[1];      // int64 per reference
  const unsigned char* drop = (const unsigned char*)d_in[2];  // bool per reference
  // d_in[3] num_runs, d_in[4] p_percent: compile-time constants here
  const float* W1  = (const float*)d_in[5];
  const float* b1  = (const float*)d_in[6];
  const float* Wg  = (const float*)d_in[7];
  const float* a_s = (const float*)d_in[8];
  const float* a_d = (const float*)d_in[9];
  const float* bg  = (const float*)d_in[10];
  const float* W2  = (const float*)d_in[11];
  const float* b2  = (const float*)d_in[12];
  const float* Wm1 = (const float*)d_in[13];
  const float* bm1 = (const float*)d_in[14];
  const float* Wm2 = (const float*)d_in[15];
  const float* bm2 = (const float*)d_in[16];
  float* out = (float*)d_out;

  // -------- workspace carve (256B aligned), total ~43 MB --------
  char* base = (char*)d_ws; size_t off = 0;
  auto carve = [&](size_t bytes) -> void* {
    off = (off + 255) & ~(size_t)255;
    void* p = base + off; off += bytes; return p;
  };
  float*     mean = (float*)carve(kF * 4);
  float*     rstd = (float*)carve(kF * 4);
  unsigned*  mmin = (unsigned*)carve(kR * 4);
  unsigned*  mmax = (unsigned*)carve(kR * 4);
  unsigned*  thr  = (unsigned*)carve(kR * 3 * 4);
  unsigned*  cnt  = (unsigned*)carve(kR * 2 * 4);
  unsigned*  hist = (unsigned*)carve((size_t)kR * kBIN * 4);
  _Float16*  W1c  = (_Float16*)carve(64 * 32 * 2);
  _Float16*  Wgc  = (_Float16*)carve(64 * 64 * 2);
  _Float16*  W2c  = (_Float16*)carve(16 * 64 * 2);
  _Float16*  Wm1c = (_Float16*)carve(64 * 32 * 2);
  _Float16*  xrh  = (_Float16*)carve((size_t)kNB * kF * 2);   // 2.56 MB
  float*     sbuf = (float*)carve((size_t)kRE * 4);           // 5.12 MB
  unsigned*  keyu = (unsigned*)carve((size_t)kRE * 4);        // 5.12 MB
  int*       sel  = (int*)carve((size_t)kR * kNRES * 4);      // 3.58 MB
  float*     deg  = (float*)carve((size_t)kNB * 4);
  float*     dinv = (float*)carve((size_t)kNB * 4);
  float*     xw   = (float*)carve((size_t)kNB * kH * 4);      // 10.24 MB
  float*     agg  = (float*)carve((size_t)kNB * kH * 4);      // 10.24 MB
  _Float16*  hh   = (_Float16*)carve((size_t)kNB * kH * 2);   // 5.12 MB
  float*     es   = (float*)carve((size_t)kNB * 4);
  float*     edv  = (float*)carve((size_t)kNB * 4);
  unsigned*  mU   = (unsigned*)carve((size_t)kNB * 4);
  float*     zbuf = (float*)carve((size_t)kNB * 4);

  const int B = 256;
  dim3 blk(B);
  auto blocks = [](long long work, int b) { return dim3((unsigned)((work + b - 1) / b)); };

  // ---- stage 0: standardize + build f16 xr ----
  psagnn_colstats<<<dim3(kF), blk, 0, stream>>>(x, mean, rstd);
  psagnn_build_xr<<<blocks((long long)kNB * kF, B), blk, 0, stream>>>(x, drop, mean, rstd, xrh);
  psagnn_prep_weights<<<dim3(1), blk, 0, stream>>>(W1, Wg, W2, Wm1, W1c, Wgc, W2c, Wm1c);

  // ---- stage 1: fused edge MLP (WMMA), 80000 waves ----
  psagnn_edge_mlp_wmma<<<dim3((kR * (kE / 16)) / (B / 32)), blk, 0, stream>>>(
      xrh, ei, Wm1c, bm1, Wm2, bm2, sbuf);

  // ---- stage 2: min/max, keys, histogram threshold top-k ----
  psagnn_init_control<<<dim3(16), blk, 0, stream>>>(mmin, mmax, hist);
  psagnn_minmax<<<dim3(2048), blk, 0, stream>>>(sbuf, mmin, mmax);
  psagnn_key_hist<<<dim3(2048), blk, 0, stream>>>(sbuf, mmin, mmax, keyu, hist);
  psagnn_threshold<<<dim3(kR), dim3(32), 0, stream>>>(hist, thr, cnt);
  psagnn_select<<<dim3(2048), blk, 0, stream>>>(keyu, thr, cnt, sel);
  psagnn_write_sel<<<blocks((long long)kR * kNRES, B), blk, 0, stream>>>(
      sel, ei, out + (size_t)kR * kN * kL);

  // ---- stage 3: shared symmetric GCN degree ----
  psagnn_deg_init<<<blocks(kNB, B), blk, 0, stream>>>(deg);
  psagnn_deg_edge<<<dim3(5000), blk, 0, stream>>>(ei, deg);
  psagnn_dinv<<<blocks(kNB, B), blk, 0, stream>>>(deg, dinv);

  // ---- GCN1: xw = xr @ W1 (WMMA), aggregate, relu -> hh ----
  psagnn_gemm_wmma<<<blocks((long long)(kNB / 16) * (kH / 16) * 32, B), blk, 0, stream>>>(
      xrh, W1c, xw, kNB, kF, kH);
  psagnn_gcn_init<<<blocks((long long)kNB * kH, B), blk, 0, stream>>>(xw, b1, dinv, agg, kH, kH);
  psagnn_gcn_edge<<<dim3(32768), blk, 0, stream>>>(ei, xw, dinv, agg, 6);
  psagnn_relu_h16<<<blocks((long long)kNB * kH, B), blk, 0, stream>>>(agg, hh, (long long)kNB * kH);

  // ---- 2x GAT (same weights): xw = h @ Wg (WMMA), softmax attention ----
  for (int layer = 0; layer < 2; ++layer) {
    psagnn_gemm_wmma<<<blocks((long long)(kNB / 16) * (kH / 16) * 32, B), blk, 0, stream>>>(
        hh, Wgc, xw, kNB, kH, kH);
    psagnn_attn_prep<<<blocks(kNB, B), blk, 0, stream>>>(xw, a_s, a_d, es, edv, mU);
    psagnn_gat_max<<<dim3(5000), blk, 0, stream>>>(ei, es, edv, mU);
    psagnn_gat_z_init<<<blocks(kNB, B), blk, 0, stream>>>(es, edv, mU, zbuf);
    psagnn_gat_z_edge<<<dim3(5000), blk, 0, stream>>>(ei, es, edv, mU, zbuf);
    psagnn_gat_out_init<<<blocks((long long)kNB * kH, B), blk, 0, stream>>>(
        xw, bg, es, edv, mU, zbuf, agg);
    psagnn_gat_out_edge<<<dim3(32768), blk, 0, stream>>>(ei, xw, es, edv, mU, zbuf, agg);
    psagnn_relu_h16<<<blocks((long long)kNB * kH, B), blk, 0, stream>>>(agg, hh, (long long)kNB * kH);
  }

  // ---- GCN2: xw = h @ W2pad (WMMA, Nn=16), aggregate, log_softmax -> out ----
  psagnn_gemm_wmma<<<blocks((long long)(kNB / 16) * (kLP / 16) * 32, B), blk, 0, stream>>>(
      hh, W2c, xw, kNB, kH, kLP);
  psagnn_gcn_init<<<blocks((long long)kNB * kLP, B), blk, 0, stream>>>(xw, b2, dinv, agg, kLP, kL);
  psagnn_gcn_edge<<<dim3(16384), blk, 0, stream>>>(ei, xw, dinv, agg, 4);
  psagnn_logsoftmax<<<blocks(kNB, B), blk, 0, stream>>>(agg, out);
}
